// SlowMLP_1331439862247
// MI455X (gfx1250) — matrix operand order repaired and checked by reading, not compiled, not run on hardware
//
#include <hip/hip_runtime.h>
#include <stdint.h>

// ---------------------------------------------------------------------------
// Types for the CDNA5 WMMA builtin: D(f32 16x16) = A(bf16 16x32) x B(bf16 32x16) + C
// ---------------------------------------------------------------------------
typedef __attribute__((ext_vector_type(16))) __bf16 bf16x16;
typedef __attribute__((ext_vector_type(8)))  float  f32x8;

union FragU {
    bf16x16 v;
    uint4   q[2];   // two 16-byte chunks per lane
};

// Round-to-nearest-even fp32 -> bf16 (bit pattern), no dependence on __bf16 arithmetic.
__device__ __forceinline__ unsigned short f2bf(float f) {
    union { float f; unsigned int u; } c; c.f = f;
    unsigned int u = c.u;
    u += 0x7FFFu + ((u >> 16) & 1u);
    return (unsigned short)(u >> 16);
}

// ---------------------------------------------------------------------------
// Kernel 1: elementwise fp32 -> bf16 (8 elements / thread, 16B stores)
// ---------------------------------------------------------------------------
__global__ void cvt_f32_to_bf16(const float* __restrict__ src,
                                unsigned short* __restrict__ dst, size_t n) {
    size_t base = ((size_t)blockIdx.x * blockDim.x + threadIdx.x) * 8;
    if (base >= n) return;
    float4 f0 = *(const float4*)(src + base);
    float4 f1 = *(const float4*)(src + base + 4);
    union { unsigned short h[8]; uint4 q; } o;
    o.h[0] = f2bf(f0.x); o.h[1] = f2bf(f0.y); o.h[2] = f2bf(f0.z); o.h[3] = f2bf(f0.w);
    o.h[4] = f2bf(f1.x); o.h[5] = f2bf(f1.y); o.h[6] = f2bf(f1.z); o.h[7] = f2bf(f1.w);
    *(uint4*)(dst + base) = o.q;
}

// ---------------------------------------------------------------------------
// Kernel 2: LDS-tiled transpose + convert: dst[c*R + r] = bf16(src[r*C + c])
// block = (32,8), grid = (C/32, R/32). R,C multiples of 32.
// ---------------------------------------------------------------------------
__global__ void transpose_to_bf16(const float* __restrict__ src,
                                  unsigned short* __restrict__ dst, int R, int C) {
    __shared__ float tile[32][33];
    const int tx = threadIdx.x, ty = threadIdx.y;
    const int r0 = blockIdx.y * 32, c0 = blockIdx.x * 32;
#pragma unroll
    for (int j = 0; j < 4; ++j)
        tile[ty + 8 * j][tx] = src[(size_t)(r0 + ty + 8 * j) * C + (c0 + tx)];
    __syncthreads();
#pragma unroll
    for (int j = 0; j < 4; ++j)
        dst[(size_t)(c0 + ty + 8 * j) * R + (r0 + tx)] = f2bf(tile[tx][ty + 8 * j]);
}

// ---------------------------------------------------------------------------
// Kernel 3: GEMM  out = act(A @ Bt^T + bias)
//   A : [M,K] bf16 row-major, Bt : [N,K] bf16 row-major (i.e. B transposed)
//   wave32: each wave computes a 64x64 tile as 4x4 WMMA 16x16 tiles.
//   8 waves/WG -> 128(M) x 256(N) workgroup tile. No LDS: L2 (192MB) holds
//   the full bf16 operand sets, so direct global b128 loads feed WMMA.
// ---------------------------------------------------------------------------
template <bool RELU, bool OUT_BF16>
__global__ void __launch_bounds__(256)
wmma_gemm_bias_act(const unsigned short* __restrict__ A,
                   const unsigned short* __restrict__ Bt,
                   const float* __restrict__ bias,
                   void* __restrict__ out,
                   int N, int K) {
    const int lane = threadIdx.x & 31;
    const int wv   = threadIdx.x >> 5;
    const int wm   = wv & 1;           // 2 waves along M
    const int wn   = wv >> 1;          // 4 waves along N
    const int mBase = blockIdx.y * 128 + wm * 64;
    const int nBase = blockIdx.x * 256 + wn * 64;
    const int lr = lane & 15;          // A: row-in-tile, B: col-in-tile
    const int hi = lane >> 4;          // lane half selects K sub-range

    // Per-lane base pointers, K layouts per CDNA5 ISA 7.12.2:
    //  A 16-bit 16x32: lanes 0-15 hold K{0..7,16..23}; lanes 16-31 K{8..15,24..31}
    //  B 16-bit 32x16: lanes 0-15 hold K{0..7,16..23}; lanes 16-31 K{8..15,24..31}
    const unsigned short* aptr[4];
    const unsigned short* bptr[4];
#pragma unroll
    for (int i = 0; i < 4; ++i)
        aptr[i] = A + (size_t)(mBase + 16 * i + lr) * K + hi * 8;
#pragma unroll
    for (int j = 0; j < 4; ++j)
        bptr[j] = Bt + (size_t)(nBase + 16 * j + lr) * K + hi * 8;

    f32x8 acc[4][4] = {};

    for (int k = 0; k < K; k += 32) {
        FragU a[4], b[4];
#pragma unroll
        for (int i = 0; i < 4; ++i) {
            a[i].q[0] = *(const uint4*)(aptr[i] + k);        // K +0..7  (or +8..15)
            a[i].q[1] = *(const uint4*)(aptr[i] + k + 16);   // K +16..23 (or +24..31)
        }
#pragma unroll
        for (int j = 0; j < 4; ++j) {
            b[j].q[0] = *(const uint4*)(bptr[j] + k);        // K +0..7  (or +8..15)
            b[j].q[1] = *(const uint4*)(bptr[j] + k + 16);   // K +16..23 (or +24..31)
        }
#pragma unroll
        for (int i = 0; i < 4; ++i)
#pragma unroll
            for (int j = 0; j < 4; ++j)
                acc[i][j] = __builtin_amdgcn_wmma_f32_16x16x32_bf16(
                    /*neg_a=*/false, a[i].v, /*neg_b=*/false, b[j].v,
                    /*c_mod=*/(short)0, acc[i][j],
                    /*reuse_a=*/false, /*reuse_b=*/false);
    }

    // Epilogue. C/D layout: VGPR r, lanes 0-15 -> row r, lanes 16-31 -> row r+8.
#pragma unroll
    for (int j = 0; j < 4; ++j) {
        const int col = nBase + 16 * j + lr;
        const float bj = bias[col];
#pragma unroll
        for (int i = 0; i < 4; ++i) {
            const int rowBase = mBase + 16 * i + 8 * hi;
#pragma unroll
            for (int r = 0; r < 8; ++r) {
                float v = acc[i][j][r] + bj;
                if (RELU) v = v > 0.0f ? v : 0.0f;
                if constexpr (OUT_BF16)
                    ((unsigned short*)out)[(size_t)(rowBase + r) * N + col] = f2bf(v);
                else
                    ((float*)out)[(size_t)(rowBase + r) * N + col] = v;
            }
        }
    }
}

// ---------------------------------------------------------------------------
// Launch: convert -> transpose-convert weights -> GEMM1(relu, bf16 out) -> GEMM2
// Workspace: xb 32MB | w1t 32MB | w2t 32MB | h_bf16 128MB  (224MB total)
// ---------------------------------------------------------------------------
extern "C" void kernel_launch(void* const* d_in, const int* in_sizes, int n_in,
                              void* d_out, int out_size, void* d_ws, size_t ws_size,
                              hipStream_t stream) {
    (void)in_sizes; (void)n_in; (void)out_size; (void)ws_size;

    constexpr int Bm = 8192, Din = 2048, Dh = 8192, Dout = 2048;

    const float* x  = (const float*)d_in[0];   // [Bm, Din]
    const float* w1 = (const float*)d_in[1];   // [Din, Dh]
    const float* b1 = (const float*)d_in[2];   // [Dh]
    const float* w2 = (const float*)d_in[3];   // [Dh, Dout]
    const float* b2 = (const float*)d_in[4];   // [Dout]
    float* out = (float*)d_out;                // [Bm, Dout]

    char* ws = (char*)d_ws;
    const size_t MB = 1024ull * 1024ull;
    unsigned short* xb  = (unsigned short*)(ws);             // [Bm,  Din] bf16
    unsigned short* w1t = (unsigned short*)(ws + 32 * MB);   // [Dh,  Din] bf16 (w1^T)
    unsigned short* w2t = (unsigned short*)(ws + 64 * MB);   // [Dout,Dh ] bf16 (w2^T)
    unsigned short* hb  = (unsigned short*)(ws + 96 * MB);   // [Bm,  Dh ] bf16

    // x -> bf16 (16.7M elems, 8/thread)
    cvt_f32_to_bf16<<<(Bm * (size_t)Din) / (8 * 256), 256, 0, stream>>>(
        x, xb, (size_t)Bm * Din);

    // w1 [Din,Dh] -> w1t [Dh,Din]
    transpose_to_bf16<<<dim3(Dh / 32, Din / 32), dim3(32, 8), 0, stream>>>(
        w1, w1t, Din, Dh);
    // w2 [Dh,Dout] -> w2t [Dout,Dh]
    transpose_to_bf16<<<dim3(Dout / 32, Dh / 32), dim3(32, 8), 0, stream>>>(
        w2, w2t, Dh, Dout);

    // h = relu(x @ w1 + b1), bf16 out
    wmma_gemm_bias_act<true, true>
        <<<dim3(Dh / 256, Bm / 128), 256, 0, stream>>>(xb, w1t, b1, hb, Dh, Din);

    // out = h @ w2 + b2, fp32 out
    wmma_gemm_bias_act<false, false>
        <<<dim3(Dout / 256, Bm / 128), 256, 0, stream>>>(hb, w2t, b2, out, Dout, Dh);
}
